// GeoSlot_19550691131569
// MI455X (gfx1250) — matrix operand order repaired and checked by reading, not compiled, not run on hardware
//
#include <hip/hip_runtime.h>
#include <hip/hip_bf16.h>

typedef unsigned short u16;
typedef unsigned int   u32;

typedef __attribute__((ext_vector_type(16))) __bf16 v16bf;
typedef __attribute__((ext_vector_type(8)))  float  f32x8;

// ---------------- constants ----------------
#define BB   32
#define NN   1024
#define TOK  32768      // BB*NN
#define FD   1568
#define SD   256
#define TS   28         // MAX_SLOTS + N_REGISTER
#define MS   24
#define NH   4
#define DHD  64
#define SHN  112        // TS*NH

// output layout (floats)
#define OBJ_OFF   0
#define REG_OFF   196608
#define MASK_OFF  229376
#define ATTN_OFF  262144
#define DEC_OFF   1179648
#define KEEP_OFF  1180416

// ---------------- helpers ----------------
__device__ inline u16 f2bf(float f) {
    u32 u = __float_as_uint(f);
    u += 0x7fffu + ((u >> 16) & 1u);        // round-to-nearest-even
    return (u16)(u >> 16);
}
__device__ inline float bf2f(u16 h) {
    return __uint_as_float(((u32)h) << 16);
}

union Frag { v16bf v; uint4 q[2]; };

__device__ inline f32x8 zero8() {
    f32x8 z = {0.f,0.f,0.f,0.f,0.f,0.f,0.f,0.f};
    return z;
}

__device__ inline f32x8 wmma_bf16(const Frag& a, const Frag& b, f32x8 c) {
    return __builtin_amdgcn_wmma_f32_16x16x32_bf16(false, a.v, false, b.v,
                                                   (short)0, c, false, false);
}

// A fragment (16x32 bf16): per lane M=lane%16, g=lane/16.
// VGPR r<4 : K = 8g + 2r + lo ; VGPR r>=4 : K = 16 + 8g + 2(r-4) + lo
__device__ inline Frag ldFragA(const uint4* lds, int row, int rowStride4, int kc8, int g) {
    Frag f; const uint4* p = lds + row * rowStride4 + kc8;
    f.q[0] = p[g];
    f.q[1] = p[2 + g];
    return f;
}
// B fragment (32x16 bf16): per lane N=lane%16, g=lane/16.
// VGPR r<4 : K = 8g + 2r + lo ; VGPR r>=4 : K = 16 + 8g + 2(r-4) + lo
__device__ inline Frag ldFragB(const uint4* lds, int row, int rowStride4, int kc8, int g) {
    Frag f; const uint4* p = lds + row * rowStride4 + kc8;
    f.q[0] = p[g];
    f.q[1] = p[2 + g];
    return f;
}

__device__ inline float blockSum(float v, float* red) {
    red[threadIdx.x] = v; __syncthreads();
    for (int st = 128; st > 0; st >>= 1) {
        if ((int)threadIdx.x < st) red[threadIdx.x] += red[threadIdx.x + st];
        __syncthreads();
    }
    float r = red[0]; __syncthreads();
    return r;
}

__device__ inline float sigm(float x) { return 1.f / (1.f + expf(-x)); }

// ---------------- weight conversion (f32 [K][N] -> bf16 [N][K] transposed) ----------------
__global__ __launch_bounds__(256) void k_convT(const float* __restrict__ W,
                                               u16* __restrict__ Wt,
                                               int K, int N, int ldK, int nOff) {
    int i = blockIdx.x * 256 + threadIdx.x;
    if (i >= K * N) return;
    int k = i / N, n = i % N;
    Wt[(size_t)(n + nOff) * ldK + k] = f2bf(W[i]);
}

// hi/lo split variant (for the big feature GEMM weights)
__global__ __launch_bounds__(256) void k_convT2(const float* __restrict__ W,
                                                u16* __restrict__ Wh,
                                                u16* __restrict__ Wl,
                                                int K, int N, int ldK, int nOff) {
    int i = blockIdx.x * 256 + threadIdx.x;
    if (i >= K * N) return;
    int k = i / N, n = i % N;
    float w = W[i];
    u16 h = f2bf(w);
    size_t idx = (size_t)(n + nOff) * ldK + k;
    Wh[idx] = h;
    Wl[idx] = f2bf(w - bf2f(h));
}

// ---------------- fused feature GEMM (bf16 hi/lo split) + bg mask + proj + LN ----------------
// block: 32 tokens x 320 outputs, K=1568
__global__ __launch_bounds__(256) void k_feat(const float* __restrict__ feat,
                                              const u16* __restrict__ wcath,  // [320][1568]
                                              const u16* __restrict__ wcatl,  // [320][1568]
                                              const float* __restrict__ bg_b1,
                                              const float* __restrict__ bg_w2,
                                              const float* __restrict__ bg_b2,
                                              const float* __restrict__ proj_b,
                                              const float* __restrict__ g_in,
                                              const float* __restrict__ b_in,
                                              u16* __restrict__ inputs_bf,
                                              float* __restrict__ out_mask) {
    __shared__ __align__(16) char smem[45056];
    u16* lAh = (u16*)smem;              //  2048 B
    u16* lAl = (u16*)(smem + 2048);     //  2048 B
    u16* lBh = (u16*)(smem + 4096);     // 20480 B
    u16* lBl = (u16*)(smem + 24576);    // 20480 B
    // epilogue aliases (valid only after the K loop):
    float* lY      = (float*)smem;            // 32*320*4 = 40960 B
    float* red     = (float*)(smem + 40960);  // 1024 B
    float* redv    = (float*)(smem + 41984);  // 1024 B
    float* tokscal = (float*)(smem + 43008);  // 128 B

    int tid = threadIdx.x;
    int lane = tid & 31, wid = tid >> 5;
    int lrow = lane & 15, g = lane >> 4;
    int m0 = blockIdx.x * 32;

    f32x8 acc[5];
    for (int t = 0; t < 5; t++) acc[t] = zero8();

    int ar = tid >> 3;            // A-staging row 0..31
    int ac = tid & 7;             // float4 column 0..7

    for (int k0 = 0; k0 < FD; k0 += 32) {
        // A: 32x32 f32 -> hi/lo bf16 (one float4 per thread, B128 load)
        {
            float4 f = *(const float4*)(feat + (size_t)(m0 + ar) * FD + k0 + ac * 4);
            u16 hx = f2bf(f.x), hy = f2bf(f.y), hz = f2bf(f.z), hw = f2bf(f.w);
            u16 lx = f2bf(f.x - bf2f(hx)), ly = f2bf(f.y - bf2f(hy));
            u16 lz = f2bf(f.z - bf2f(hz)), lw = f2bf(f.w - bf2f(hw));
            u32* Ah32 = (u32*)lAh; u32* Al32 = (u32*)lAl;
            Ah32[ar * 16 + ac * 2]     = (u32)hx | ((u32)hy << 16);
            Ah32[ar * 16 + ac * 2 + 1] = (u32)hz | ((u32)hw << 16);
            Al32[ar * 16 + ac * 2]     = (u32)lx | ((u32)ly << 16);
            Al32[ar * 16 + ac * 2 + 1] = (u32)lz | ((u32)lw << 16);
        }
        // B: 320x32 bf16 hi + lo (uint4 copies)
        for (int i = tid; i < 1280; i += 256) {
            int r = i >> 2, q = i & 3;
            ((uint4*)lBh)[i] = *(const uint4*)(wcath + (size_t)r * FD + k0 + q * 8);
            ((uint4*)lBl)[i] = *(const uint4*)(wcatl + (size_t)r * FD + k0 + q * 8);
        }
        // prefetch next K chunk of features (global_prefetch_b8 path)
        if (k0 + 32 < FD)
            __builtin_prefetch(feat + (size_t)(m0 + ar) * FD + (k0 + 32) + ac * 4, 0, 0);
        __syncthreads();

        const uint4* Ah4 = (const uint4*)lAh;
        const uint4* Al4 = (const uint4*)lAl;
        const uint4* Bh4 = (const uint4*)lBh;
        const uint4* Bl4 = (const uint4*)lBl;
        for (int t = 0; t < 5; t++) {
            int ti = wid * 5 + t;
            int mi = ti & 1, ni = ti >> 1;
            Frag afh = ldFragA(Ah4, mi * 16 + lrow, 4, 0, g);
            Frag afl = ldFragA(Al4, mi * 16 + lrow, 4, 0, g);
            Frag bfh = ldFragB(Bh4, ni * 16 + lrow, 4, 0, g);
            Frag bfl = ldFragB(Bl4, ni * 16 + lrow, 4, 0, g);
            acc[t] = wmma_bf16(afh, bfh, acc[t]);   // hi*hi
            acc[t] = wmma_bf16(afl, bfh, acc[t]);   // lo*hi
            acc[t] = wmma_bf16(afh, bfl, acc[t]);   // hi*lo
        }
        __syncthreads();
    }
    // dump accumulators to LDS tile (aliases staging buffers; all WMMA done)
    for (int t = 0; t < 5; t++) {
        int ti = wid * 5 + t;
        int mi = ti & 1, ni = ti >> 1;
        int col = ni * 16 + lrow;
        for (int r = 0; r < 8; r++) {
            int row = mi * 16 + r + 8 * g;
            lY[row * 320 + col] = acc[t][r];
        }
    }
    __syncthreads();

    // epilogue: 8 threads per token
    int tk = tid >> 3;
    int l8 = tid & 7;

    float part = 0.f;
    for (int j = l8; j < 64; j += 8) {
        float hv = lY[tk * 320 + j] + bg_b1[j];
        hv = hv > 0.f ? hv : 0.f;
        part += hv * bg_w2[j];
    }
    red[tid] = part; __syncthreads();
    if (l8 == 0) {
        float s = 0.f;
        for (int i = 0; i < 8; i++) s += red[tk * 8 + i];
        s += bg_b2[0];
        float mask = sigm(s);
        tokscal[tk] = mask;
        out_mask[m0 + tk] = mask;
    }
    __syncthreads();
    float mask = tokscal[tk];

    float ps = 0.f, ps2 = 0.f;
    for (int j = l8; j < 256; j += 8) {
        float fp = mask * lY[tk * 320 + 64 + j] + proj_b[j];
        lY[tk * 320 + 64 + j] = fp;
        ps += fp; ps2 += fp * fp;
    }
    red[tid] = ps; redv[tid] = ps2; __syncthreads();
    if (l8 == 0) {
        float s = 0.f, s2 = 0.f;
        for (int i = 0; i < 8; i++) { s += red[tk * 8 + i]; s2 += redv[tk * 8 + i]; }
        float mean = s * (1.f / 256.f);
        float var  = s2 * (1.f / 256.f) - mean * mean;
        red[tk * 8]  = mean;
        redv[tk * 8] = rsqrtf(var + 1e-5f);
    }
    __syncthreads();
    float mean = red[tk * 8], inv = redv[tk * 8];
    for (int j = l8; j < 256; j += 8) {
        float x = (lY[tk * 320 + 64 + j] - mean) * inv * g_in[j] + b_in[j];
        inputs_bf[(size_t)(m0 + tk) * 256 + j] = f2bf(x);
    }
}

// ---------------- generic bf16 GEMM: C[M,N] = act(A[M,K] @ Wt[N,K]^T + bias), bf16 out ----------
__global__ __launch_bounds__(256) void k_gemm_bf(const u16* __restrict__ A,
                                                 const u16* __restrict__ Wt,
                                                 const float* __restrict__ bias,
                                                 u16* __restrict__ C,
                                                 int M, int N, int K, int act) {
    __shared__ __align__(16) u16 lA[64 * 32];
    __shared__ __align__(16) u16 lB[64 * 32];
    int tid = threadIdx.x, lane = tid & 31, wid = tid >> 5;
    int lrow = lane & 15, g = lane >> 4;
    int m0 = blockIdx.x * 64, n0 = blockIdx.y * 64;
    f32x8 acc0 = zero8(), acc1 = zero8();
    int ti0 = wid * 2;
    int sr = tid >> 2, sq = tid & 3;   // staging: one uint4 per thread per matrix

    for (int k0 = 0; k0 < K; k0 += 32) {
        ((uint4*)lA)[tid] = *(const uint4*)(A  + (size_t)(m0 + sr) * K + k0 + sq * 8);
        ((uint4*)lB)[tid] = *(const uint4*)(Wt + (size_t)(n0 + sr) * K + k0 + sq * 8);
        __syncthreads();
        {
            int mi = ti0 & 3, ni = ti0 >> 2;
            Frag af = ldFragA((const uint4*)lA, mi * 16 + lrow, 4, 0, g);
            Frag bf = ldFragB((const uint4*)lB, ni * 16 + lrow, 4, 0, g);
            acc0 = wmma_bf16(af, bf, acc0);
        }
        {
            int ti = ti0 + 1;
            int mi = ti & 3, ni = ti >> 2;
            Frag af = ldFragA((const uint4*)lA, mi * 16 + lrow, 4, 0, g);
            Frag bf = ldFragB((const uint4*)lB, ni * 16 + lrow, 4, 0, g);
            acc1 = wmma_bf16(af, bf, acc1);
        }
        __syncthreads();
    }
    for (int t = 0; t < 2; t++) {
        int ti = ti0 + t;
        int mi = ti & 3, ni = ti >> 2;
        f32x8& acc = t ? acc1 : acc0;
        int n = n0 + ni * 16 + lrow;
        float bv = bias ? bias[n] : 0.f;
        for (int r = 0; r < 8; r++) {
            int m = m0 + mi * 16 + r + 8 * g;
            float v = acc[r] + bv;
            if (act == 1) v = 0.5f * v * (1.f + erff(v * 0.70710678118654752f)); // exact GELU
            C[(size_t)m * N + n] = f2bf(v);
        }
    }
}

// ---------------- FF2 GEMM: out = res + A @ Wt^T + bias (f32 out, N=256) ----------------
__global__ __launch_bounds__(256) void k_gemm_ff2(const u16* __restrict__ A,
                                                  const u16* __restrict__ Wt,
                                                  const float* __restrict__ bias,
                                                  const float* __restrict__ res,
                                                  float* __restrict__ out,
                                                  int M, int N, int K) {
    __shared__ __align__(16) u16 lA[64 * 32];
    __shared__ __align__(16) u16 lB[64 * 32];
    int tid = threadIdx.x, lane = tid & 31, wid = tid >> 5;
    int lrow = lane & 15, g = lane >> 4;
    int m0 = blockIdx.x * 64, n0 = blockIdx.y * 64;
    f32x8 acc0 = zero8(), acc1 = zero8();
    int ti0 = wid * 2;
    int sr = tid >> 2, sq = tid & 3;

    for (int k0 = 0; k0 < K; k0 += 32) {
        ((uint4*)lA)[tid] = *(const uint4*)(A  + (size_t)(m0 + sr) * K + k0 + sq * 8);
        ((uint4*)lB)[tid] = *(const uint4*)(Wt + (size_t)(n0 + sr) * K + k0 + sq * 8);
        __syncthreads();
        for (int t = 0; t < 2; t++) {
            int ti = ti0 + t;
            int mi = ti & 3, ni = ti >> 2;
            Frag af = ldFragA((const uint4*)lA, mi * 16 + lrow, 4, 0, g);
            Frag bf = ldFragB((const uint4*)lB, ni * 16 + lrow, 4, 0, g);
            if (t == 0) acc0 = wmma_bf16(af, bf, acc0);
            else        acc1 = wmma_bf16(af, bf, acc1);
        }
        __syncthreads();
    }
    for (int t = 0; t < 2; t++) {
        int ti = ti0 + t;
        int mi = ti & 3, ni = ti >> 2;
        f32x8& acc = t ? acc1 : acc0;
        int n = n0 + ni * 16 + lrow;
        float bv = bias[n];
        for (int r = 0; r < 8; r++) {
            int m = m0 + mi * 16 + r + 8 * g;
            out[(size_t)m * N + n] = res[(size_t)m * N + n] + acc[r] + bv;
        }
    }
}

// ---------------- slot init ----------------
__global__ __launch_bounds__(256) void k_slots_init(const float* __restrict__ mu,
                                                    const float* __restrict__ ls,
                                                    const float* __restrict__ noise,
                                                    float* __restrict__ slots) {
    int i = blockIdx.x * 256 + threadIdx.x;      // 0..229375
    int j = i % (TS * SD);
    slots[i] = mu[j] + expf(ls[j]) * noise[i];
}

// ---------------- per-row LayerNorm (256 cols) -> bf16 ----------------
__global__ __launch_bounds__(256) void k_rowln(const float* __restrict__ x,
                                               const float* __restrict__ gw,
                                               const float* __restrict__ bw,
                                               u16* __restrict__ y) {
    __shared__ float red[256];
    int row = blockIdx.x, tid = threadIdx.x;
    float v = x[(size_t)row * 256 + tid];
    float mean = blockSum(v, red) * (1.f / 256.f);
    float d = v - mean;
    float var = blockSum(d * d, red) * (1.f / 256.f);
    float o = d * rsqrtf(var + 1e-5f) * gw[tid] + bw[tid];
    y[(size_t)row * 256 + tid] = f2bf(o);
}

// ---------------- attention: dots + joint softmax + attn_out ----------------
// grid (B, N/64)
__global__ __launch_bounds__(256) void k_attn(const u16* __restrict__ q_bf,
                                              const u16* __restrict__ kv_bf,
                                              float* __restrict__ attn_f,
                                              float* __restrict__ attn_out) {
    __shared__ __align__(16) u16 lQ[4 * 32 * 64];
    __shared__ __align__(16) u16 lK[64 * 64];
    __shared__ float lD[112 * 64];

    int b = blockIdx.x, n0 = blockIdx.y * 64;
    int tid = threadIdx.x, lane = tid & 31, wid = tid >> 5;
    int lrow = lane & 15, g = lane >> 4;

    // stage q (uint4): 4 heads x 32 rows x 8 uint4
    for (int i = tid; i < 1024; i += 256) {
        int h = i >> 8, s = (i >> 3) & 31, q = i & 7;
        uint4 v4; v4.x = v4.y = v4.z = v4.w = 0u;
        if (s < TS)
            v4 = *(const uint4*)(q_bf + (size_t)(b * TS + s) * 256 + h * 64 + q * 8);
        ((uint4*)lQ)[i] = v4;
    }
    int mi = wid & 1, ni = wid >> 1;   // ni 0..3
    for (int h = 0; h < 4; h++) {
        // stage k for head h (uint4): 64 rows x 8 uint4
        for (int i = tid; i < 512; i += 256) {
            int t = i >> 3, q = i & 7;
            ((uint4*)lK)[i] =
                *(const uint4*)(kv_bf + (size_t)(b * NN + n0 + t) * 512 + h * 64 + q * 8);
        }
        __syncthreads();
        f32x8 acc = zero8();
        const uint4* Q4 = (const uint4*)(lQ + h * 32 * 64);
        const uint4* K4 = (const uint4*)lK;
        for (int kc = 0; kc < 64; kc += 32) {
            Frag af = ldFragA(Q4, mi * 16 + lrow, 8, kc >> 3, g);
            Frag bf = ldFragB(K4, ni * 16 + lrow, 8, kc >> 3, g);
            acc = wmma_bf16(af, bf, acc);
        }
        for (int r = 0; r < 8; r++) {
            int s = mi * 16 + r + 8 * g;
            if (s < TS) lD[(s * 4 + h) * 64 + ni * 16 + lrow] = acc[r] * 0.125f; // DH^-0.5
        }
        __syncthreads();
    }
    // softmax over 112 slot-heads per token column
    if (tid < 64) {
        int c = tid;
        float mx = -3.4e38f;
        for (int s = 0; s < SHN; s++) mx = fmaxf(mx, lD[s * 64 + c]);
        float sum = 0.f;
        for (int s = 0; s < SHN; s++) {
            float e = expf(lD[s * 64 + c] - mx);
            lD[s * 64 + c] = e; sum += e;
        }
        float inv = 1.f / sum;
        for (int s = 0; s < TS; s++) {
            float a = 0.f;
            for (int h = 0; h < 4; h++) {
                float v = lD[(s * 4 + h) * 64 + c] * inv;
                lD[(s * 4 + h) * 64 + c] = v;
                a += v;
            }
            attn_out[(size_t)(b * TS + s) * NN + n0 + c] = a * 0.25f;
        }
    }
    __syncthreads();
    for (int i = tid; i < SHN * 64; i += 256) {
        int s = i >> 6, c = i & 63;
        attn_f[(size_t)(b * SHN + s) * NN + n0 + c] = lD[i];
    }
}

// ---------------- attn row sums ----------------
__global__ __launch_bounds__(256) void k_rowsum(const float* __restrict__ attn,
                                                float* __restrict__ rowsum) {
    __shared__ float red[256];
    int row = blockIdx.x, tid = threadIdx.x;
    float4 f = *(const float4*)(attn + (size_t)row * NN + tid * 4);
    float s = (f.x + f.y) + (f.z + f.w);
    s = blockSum(s, red);
    if (tid == 0) rowsum[row] = s;
}

// ---------------- renormalize + convert to bf16 (4 elements/thread) ----------------
__global__ __launch_bounds__(256) void k_renorm(const float* __restrict__ attn,
                                                const float* __restrict__ rowsum,
                                                u16* __restrict__ attn_bf) {
    int i4 = (blockIdx.x * 256 + threadIdx.x) * 4;
    float inv = 1.f / (rowsum[i4 >> 10] + 1e-8f);
    float4 f = *(const float4*)(attn + i4);
    u32 lo = (u32)f2bf((f.x + 1e-8f) * inv) | ((u32)f2bf((f.y + 1e-8f) * inv) << 16);
    u32 hi = (u32)f2bf((f.z + 1e-8f) * inv) | ((u32)f2bf((f.w + 1e-8f) * inv) << 16);
    uint2 v; v.x = lo; v.y = hi;
    *(uint2*)(attn_bf + i4) = v;
}

// ---------------- updates = attn @ v  (per b,h : 28x1024 @ 1024x64) ----------------
// grid (B, NH)
__global__ __launch_bounds__(256) void k_updates(const u16* __restrict__ attn_bf,
                                                 const u16* __restrict__ kv_bf,
                                                 float* __restrict__ updates) {
    __shared__ __align__(16) u16 lA[32 * 32];
    __shared__ __align__(16) u16 lB[64 * 32];
    int b = blockIdx.x, h = blockIdx.y;
    int tid = threadIdx.x, lane = tid & 31, wid = tid >> 5;
    int lrow = lane & 15, g = lane >> 4;
    int mi = wid & 1, ni = wid >> 1;    // ni 0..3
    f32x8 acc = zero8();

    for (int k0 = 0; k0 < NN; k0 += 32) {
        if (tid < 128) {                       // A: 32 rows x 4 uint4
            int s = tid >> 2, q = tid & 3;
            uint4 v4; v4.x = v4.y = v4.z = v4.w = 0u;
            if (s < TS)
                v4 = *(const uint4*)(attn_bf + (size_t)(b * SHN + s * 4 + h) * NN + k0 + q * 8);
            ((uint4*)lA)[tid] = v4;
        }
        for (int i = tid; i < 64 * 32; i += 256) {   // B: v transposed into [d][tok]
            int t = i >> 6, d = i & 63;
            lB[d * 32 + t] = kv_bf[(size_t)(b * NN + k0 + t) * 512 + 256 + h * 64 + d];
        }
        __syncthreads();
        Frag af = ldFragA((const uint4*)lA, mi * 16 + lrow, 4, 0, g);
        Frag bf = ldFragB((const uint4*)lB, ni * 16 + lrow, 4, 0, g);
        acc = wmma_bf16(af, bf, acc);
        __syncthreads();
    }
    for (int r = 0; r < 8; r++) {
        int s = mi * 16 + r + 8 * g;
        if (s < TS)
            updates[(size_t)(b * TS + s) * 256 + h * 64 + ni * 16 + lrow] = acc[r];
    }
}

// ---------------- GRU cell (896 rows) ----------------
__global__ __launch_bounds__(256) void k_gru(const float* __restrict__ up,
                                             const float* __restrict__ hprev,
                                             const float* __restrict__ wih,
                                             const float* __restrict__ whh,
                                             const float* __restrict__ bih,
                                             const float* __restrict__ bhh,
                                             float* __restrict__ outS) {
    __shared__ float su[256], sh_[256];
    int row = blockIdx.x, j = threadIdx.x;
    su[j]  = up[(size_t)row * 256 + j];
    sh_[j] = hprev[(size_t)row * 256 + j];
    __syncthreads();
    float gxr = bih[j], gxz = bih[256 + j], gxn = bih[512 + j];
    float ghr = bhh[j], ghz = bhh[256 + j], ghn = bhh[512 + j];
    const float* wr = wih + (size_t)j * 256;
    const float* wz = wih + (size_t)(256 + j) * 256;
    const float* wn = wih + (size_t)(512 + j) * 256;
    const float* vr = whh + (size_t)j * 256;
    const float* vz = whh + (size_t)(256 + j) * 256;
    const float* vn = whh + (size_t)(512 + j) * 256;
    for (int k = 0; k < 256; k++) {
        float u = su[k], h = sh_[k];
        gxr += u * wr[k]; gxz += u * wz[k]; gxn += u * wn[k];
        ghr += h * vr[k]; ghz += h * vz[k]; ghn += h * vn[k];
    }
    float r = sigm(gxr + ghr);
    float z = sigm(gxz + ghz);
    float n = tanhf(gxn + r * ghn);
    outS[(size_t)row * 256 + j] = (1.f - z) * n + z * sh_[j];
}

// ---------------- final decision head (per batch) ----------------
__global__ __launch_bounds__(256) void k_head(const float* __restrict__ slots,
                                              const float* __restrict__ gw1,
                                              const float* __restrict__ gb1,
                                              const float* __restrict__ gw2,
                                              const float* __restrict__ gb2,
                                              float* __restrict__ out) {
    __shared__ float sx[256], hid[128], redl[256];
    __shared__ float keep[24];
    __shared__ int   dec[24];
    int b = blockIdx.x, tid = threadIdx.x;

    for (int s = 0; s < MS; s++) {
        sx[tid] = slots[(size_t)(b * TS + s) * 256 + tid];
        __syncthreads();
        if (tid < 128) {
            float a = gb1[tid];
            for (int k = 0; k < 256; k++) a += sx[k] * gw1[k * 128 + tid];
            hid[tid] = a > 0.f ? a : 0.f;
        }
        __syncthreads();
        if (tid < 128) {
            redl[tid]       = hid[tid] * gw2[tid * 2 + 0];
            redl[128 + tid] = hid[tid] * gw2[tid * 2 + 1];
        }
        __syncthreads();
        for (int st = 64; st > 0; st >>= 1) {
            if (tid < st) {
                redl[tid]       += redl[tid + st];
                redl[128 + tid] += redl[128 + tid + st];
            }
            __syncthreads();
        }
        if (tid == 0) {
            float L0 = redl[0] + gb2[0];
            float L1 = redl[128] + gb2[1];
            dec[s] = (L1 > L0) ? 1 : 0;                 // argmax==1 (ties -> 0)
            float m = fmaxf(L0, L1);
            float e0 = expf(L0 - m), e1 = expf(L1 - m);
            keep[s] = e1 / (e0 + e1);
        }
        __syncthreads();
    }
    if (tid == 0) {
        int active = 0;
        for (int s = 0; s < MS; s++) active += dec[s];
        if (active < 1) {
            int am = 0; float best = keep[0];
            for (int s = 1; s < MS; s++) if (keep[s] > best) { best = keep[s]; am = s; }
            dec[am] = 1;
        }
    }
    __syncthreads();
    if (tid < MS) {
        out[DEC_OFF  + b * MS + tid] = (float)dec[tid];
        out[KEEP_OFF + b * MS + tid] = keep[tid];
    }
    for (int i = tid; i < MS * 256; i += 256) {
        int s = i >> 8, j = i & 255;
        out[OBJ_OFF + (size_t)(b * MS + s) * 256 + j] =
            slots[(size_t)(b * TS + s) * 256 + j] * (dec[s] ? 1.f : 0.f);
    }
    for (int i = tid; i < 4 * 256; i += 256) {
        int s = i >> 8, j = i & 255;
        out[REG_OFF + (size_t)(b * 4 + s) * 256 + j] =
            slots[(size_t)(b * TS + MS + s) * 256 + j];
    }
}

// ---------------- host launch ----------------
extern "C" void kernel_launch(void* const* d_in, const int* in_sizes, int n_in,
                              void* d_out, int out_size, void* d_ws, size_t ws_size,
                              hipStream_t stream) {
    const float* features = (const float*)d_in[0];
    const float* noise    = (const float*)d_in[1];
    const float* bg_w1    = (const float*)d_in[2];
    const float* bg_b1    = (const float*)d_in[3];
    const float* bg_w2    = (const float*)d_in[4];
    const float* bg_b2    = (const float*)d_in[5];
    const float* proj_w   = (const float*)d_in[6];
    const float* proj_b   = (const float*)d_in[7];
    const float* slot_mu  = (const float*)d_in[8];
    const float* slot_ls  = (const float*)d_in[9];
    const float* nin_g    = (const float*)d_in[10];
    const float* nin_b    = (const float*)d_in[11];
    const float* ns_g     = (const float*)d_in[12];
    const float* ns_b     = (const float*)d_in[13];
    const float* q_w      = (const float*)d_in[14];
    const float* k_w      = (const float*)d_in[15];
    const float* v_w      = (const float*)d_in[16];
    const float* gru_wih  = (const float*)d_in[17];
    const float* gru_whh  = (const float*)d_in[18];
    const float* gru_bih  = (const float*)d_in[19];
    const float* gru_bhh  = (const float*)d_in[20];
    const float* ffln_g   = (const float*)d_in[21];
    const float* ffln_b   = (const float*)d_in[22];
    const float* ff_w1    = (const float*)d_in[23];
    const float* ff_b1    = (const float*)d_in[24];
    const float* ff_w2    = (const float*)d_in[25];
    const float* ff_b2    = (const float*)d_in[26];
    const float* gum_w1   = (const float*)d_in[27];
    const float* gum_b1   = (const float*)d_in[28];
    const float* gum_w2   = (const float*)d_in[29];
    const float* gum_b2   = (const float*)d_in[30];
    float* out = (float*)d_out;

    char* cur = (char*)d_ws;
    auto alloc = [&](size_t bytes) -> void* {
        void* p = (void*)cur;
        cur += (bytes + 255) & ~(size_t)255;
        return p;
    };
    u16* wcath     = (u16*)alloc((size_t)320 * FD * 2);
    u16* wcatl     = (u16*)alloc((size_t)320 * FD * 2);
    u16* kvt       = (u16*)alloc((size_t)512 * 256 * 2);
    u16* qt        = (u16*)alloc((size_t)256 * 256 * 2);
    u16* ff1t      = (u16*)alloc((size_t)1024 * 256 * 2);
    u16* ff2t      = (u16*)alloc((size_t)256 * 1024 * 2);
    u16* inputs_bf = (u16*)alloc((size_t)TOK * 256 * 2);
    u16* kv_bf     = (u16*)alloc((size_t)TOK * 512 * 2);
    u16* sln_bf    = (u16*)alloc((size_t)896 * 256 * 2);
    u16* q_bf      = (u16*)alloc((size_t)896 * 256 * 2);
    u16* attn_bf   = (u16*)alloc((size_t)BB * SHN * NN * 2);
    u16* ffin_bf   = (u16*)alloc((size_t)896 * 256 * 2);
    u16* ffh_bf    = (u16*)alloc((size_t)896 * 1024 * 2);
    float* slotsA  = (float*)alloc((size_t)896 * 256 * 4);
    float* slotsB  = (float*)alloc((size_t)896 * 256 * 4);
    float* attn_f  = (float*)alloc((size_t)BB * SHN * NN * 4);
    float* rowsum  = (float*)alloc((size_t)BB * SHN * 4);
    float* updates = (float*)alloc((size_t)896 * 256 * 4);

    dim3 blk(256);
    auto cdiv = [](int a, int b) { return (a + b - 1) / b; };

    // weight conversions (f32 -> transposed bf16; feature weights hi/lo split)
    k_convT2<<<cdiv(FD * 64, 256),   blk, 0, stream>>>(bg_w1,  wcath, wcatl, FD, 64,  FD, 0);
    k_convT2<<<cdiv(FD * 256, 256),  blk, 0, stream>>>(proj_w, wcath, wcatl, FD, 256, FD, 64);
    k_convT<<<cdiv(256 * 256, 256),  blk, 0, stream>>>(k_w,  kvt, 256, 256, 256, 0);
    k_convT<<<cdiv(256 * 256, 256),  blk, 0, stream>>>(v_w,  kvt, 256, 256, 256, 256);
    k_convT<<<cdiv(256 * 256, 256),  blk, 0, stream>>>(q_w,  qt,  256, 256, 256, 0);
    k_convT<<<cdiv(256 * 1024, 256), blk, 0, stream>>>(ff_w1, ff1t, 256, 1024, 256, 0);
    k_convT<<<cdiv(1024 * 256, 256), blk, 0, stream>>>(ff_w2, ff2t, 1024, 256, 1024, 0);

    // stage 1: bg mask + projection + LN  (hi/lo split bf16 WMMA, ~fp32 accuracy)
    k_feat<<<TOK / 32, blk, 0, stream>>>(features, wcath, wcatl, bg_b1, bg_w2, bg_b2,
                                         proj_b, nin_g, nin_b, inputs_bf, out + MASK_OFF);
    // k,v projections
    k_gemm_bf<<<dim3(TOK / 64, 8), blk, 0, stream>>>(inputs_bf, kvt, nullptr, kv_bf,
                                                     TOK, 512, 256, 0);
    // slot init
    k_slots_init<<<896, blk, 0, stream>>>(slot_mu, slot_ls, noise, slotsA);

    for (int it = 0; it < 3; ++it) {
        k_rowln<<<896, blk, 0, stream>>>(slotsA, ns_g, ns_b, sln_bf);
        k_gemm_bf<<<dim3(14, 4), blk, 0, stream>>>(sln_bf, qt, nullptr, q_bf,
                                                   896, 256, 256, 0);
        k_attn<<<dim3(BB, NN / 64), blk, 0, stream>>>(q_bf, kv_bf, attn_f, out + ATTN_OFF);
        k_rowsum<<<BB * SHN, blk, 0, stream>>>(attn_f, rowsum);
        k_renorm<<<(BB * SHN * NN) / 1024, blk, 0, stream>>>(attn_f, rowsum, attn_bf);
        k_updates<<<dim3(BB, NH), blk, 0, stream>>>(attn_bf, kv_bf, updates);
        k_gru<<<896, blk, 0, stream>>>(updates, slotsA, gru_wih, gru_whh,
                                       gru_bih, gru_bhh, slotsB);
        k_rowln<<<896, blk, 0, stream>>>(slotsB, ffln_g, ffln_b, ffin_bf);
        k_gemm_bf<<<dim3(14, 16), blk, 0, stream>>>(ffin_bf, ff1t, ff_b1, ffh_bf,
                                                    896, 1024, 256, 1);
        k_gemm_ff2<<<dim3(14, 4), blk, 0, stream>>>(ffh_bf, ff2t, ff_b2, slotsB, slotsA,
                                                    896, 256, 1024);
    }
    k_head<<<BB, blk, 0, stream>>>(slotsA, gum_w1, gum_b1, gum_w2, gum_b2, out);
}
